// QuantAttnBlockINT8_41523743818205
// MI455X (gfx1250) — matrix operand order repaired and checked by reading, not compiled, not run on hardware
//
#include <hip/hip_runtime.h>

typedef __attribute__((ext_vector_type(16))) _Float16 v16h;
typedef __attribute__((ext_vector_type(8)))  _Float16 v8h;
typedef __attribute__((ext_vector_type(8)))  float    v8f;

#define B_      8
#define C_      256
#define N_      4096
#define GROUPS_ 32
#define CPG_    8
#define EPS_    1e-6f

#define WMMA_F16(A, Bm, Cc) \
  __builtin_amdgcn_wmma_f32_16x16x32_f16(false, (A), false, (Bm), (short)0, (Cc), false, false)

// Build a 16x32 f16 A-fragment for one lane: lanes 0-15 hold K(0..15),
// lanes 16-31 hold K(16..31) of row M=lane%16.
__device__ __forceinline__ v16h load_a16(const _Float16* p) {
  v8h lo = *(const v8h*)(p);
  v8h hi = *(const v8h*)(p + 8);
  v16h r;
#pragma unroll
  for (int i = 0; i < 8; ++i) { r[i] = lo[i]; r[i + 8] = hi[i]; }
  return r;
}

// Async global->LDS copy of 16 bytes per lane (CDNA5 ASYNCcnt path).
__device__ __forceinline__ void async_b128(uint32_t lds_addr, const void* gaddr) {
  uint64_t ga = (uint64_t)(uintptr_t)gaddr;
  asm volatile("global_load_async_to_lds_b128 %0, %1, off"
               :: "v"(lds_addr), "v"(ga) : "memory");
}

// ---------------------------------------------------------------------------
// GroupNorm: x[b,c,n] f32 -> hnT[b,n,c] f16 (transposed for WMMA fragments)
// ---------------------------------------------------------------------------
__global__ void gn_kernel(const float* __restrict__ x,
                          const float* __restrict__ gamma,
                          const float* __restrict__ beta,
                          _Float16* __restrict__ hnT) {
  __shared__ float red[256];
  __shared__ float red2[256];
  const int b = blockIdx.x / GROUPS_;
  const int g = blockIdx.x % GROUPS_;
  const float* xp = x + ((size_t)b * C_ + (size_t)g * CPG_) * N_;

  float s = 0.f, s2 = 0.f;
  for (int i = threadIdx.x; i < CPG_ * N_; i += 256) {
    float v = xp[i];
    s += v; s2 += v * v;
  }
  red[threadIdx.x] = s;
  red2[threadIdx.x] = s2;
  __syncthreads();
  for (int off = 128; off > 0; off >>= 1) {
    if (threadIdx.x < off) {
      red[threadIdx.x]  += red[threadIdx.x + off];
      red2[threadIdx.x] += red2[threadIdx.x + off];
    }
    __syncthreads();
  }
  const float inv_n = 1.f / (float)(CPG_ * N_);
  const float mean  = red[0] * inv_n;
  const float var   = red2[0] * inv_n - mean * mean;
  const float rstd  = rsqrtf(var + EPS_);

  for (int i = threadIdx.x; i < CPG_ * N_; i += 256) {
    const int cl = i & (CPG_ - 1);
    const int n  = i >> 3;
    const int c  = g * CPG_ + cl;
    const float v = xp[(size_t)cl * N_ + n];
    const float y = (v - mean) * rstd * gamma[c] + beta[c];
    hnT[((size_t)b * N_ + n) * C_ + c] = (_Float16)y;
  }
}

// ---------------------------------------------------------------------------
// Convert the four [256x256] f32 weights to f16 (row-major preserved)
// ---------------------------------------------------------------------------
__global__ void wcvt_kernel(const float* __restrict__ wq, const float* __restrict__ wk,
                            const float* __restrict__ wv, const float* __restrict__ wp,
                            _Float16* __restrict__ o) {
  const int i = blockIdx.x * blockDim.x + threadIdx.x;
  const int M = C_ * C_;
  if (i < M) {
    o[i]         = (_Float16)wq[i];
    o[M + i]     = (_Float16)wk[i];
    o[2 * M + i] = (_Float16)wv[i];
    o[3 * M + i] = (_Float16)wp[i];
  }
}

// ---------------------------------------------------------------------------
// QKV: qT[b,n,c] = hnT x Wq^T ; kT likewise ; v[b,c,n] = Wv x hn
// ---------------------------------------------------------------------------
__global__ void __launch_bounds__(128) qkv_kernel(
    const _Float16* __restrict__ hnT,
    const _Float16* __restrict__ wqh, const _Float16* __restrict__ wkh,
    const _Float16* __restrict__ wvh,
    const float* __restrict__ bq, const float* __restrict__ bk,
    const float* __restrict__ bv,
    _Float16* __restrict__ qT, _Float16* __restrict__ kT,
    _Float16* __restrict__ vbuf) {
  const int lane = threadIdx.x & 31;
  const int gw   = blockIdx.x * 4 + (threadIdx.x >> 5);
  const int b    = gw >> 12;
  const int r    = gw & 4095;
  const int n0   = (r >> 4) * 16;
  const int o0   = (r & 15) * 16;
  const int lm   = lane & 15;
  const int lh   = lane >> 4;

  v8f qa = {}, ka = {}, va = {};
#pragma unroll
  for (int kc = 0; kc < 8; ++kc) {
    const int c0 = kc * 32;
    v16h a_hn = load_a16(hnT + ((size_t)b * N_ + n0 + lm) * C_ + c0 + lh * 16);
    v16h b_q = *(const v16h*)(wqh + (size_t)(o0 + lm) * C_ + c0 + lh * 16);
    v16h b_k = *(const v16h*)(wkh + (size_t)(o0 + lm) * C_ + c0 + lh * 16);
    qa = WMMA_F16(a_hn, b_q, qa);
    ka = WMMA_F16(a_hn, b_k, ka);
    v16h a_wv = load_a16(wvh + (size_t)(o0 + lm) * C_ + c0 + lh * 16);
    v16h b_hn = *(const v16h*)(hnT + ((size_t)b * N_ + n0 + lm) * C_ + c0 + lh * 16);
    va = WMMA_F16(a_wv, b_hn, va);
  }
  const float bqv = bq[o0 + lm];
  const float bkv = bk[o0 + lm];
#pragma unroll
  for (int j = 0; j < 8; ++j) {
    const int n = n0 + j + lh * 8;
    const size_t idx = ((size_t)b * N_ + n) * C_ + o0 + lm;
    qT[idx] = (_Float16)(qa[j] + bqv);
    kT[idx] = (_Float16)(ka[j] + bkv);
    const int o = o0 + j + lh * 8;
    vbuf[((size_t)b * C_ + o) * N_ + n0 + lm] = (_Float16)(va[j] + bv[o]);
  }
}

// ---------------------------------------------------------------------------
// Flash attention, cooperative: 256 threads = 8 waves = 8 query tiles per
// block (one batch per block).  Each 32-key chunk of K (16KB) and V (16KB) is
// async-copied global->LDS once per block (double buffered) and consumed by
// all 8 waves, cutting L2 traffic 8x.  Online softmax as before.
// ---------------------------------------------------------------------------
__global__ void __launch_bounds__(256) attn_kernel(
    const _Float16* __restrict__ qT, const _Float16* __restrict__ kT,
    const _Float16* __restrict__ vbuf, _Float16* __restrict__ oT) {
  // K chunk: [32 keys][256 ch] halfs (rows 512B).  V chunk: [256 ch][32 keys]
  // halfs (rows 64B).  Double buffered + per-wave P staging.
  __shared__ __align__(128) _Float16 kbuf[2][32 * 256];
  __shared__ __align__(128) _Float16 vlds[2][256 * 32];
  __shared__ __align__(128) _Float16 plds_all[8][16 * 32];

  const int tid  = threadIdx.x;
  const int lane = tid & 31;
  const int w    = tid >> 5;
  _Float16* plds = plds_all[w];
  const int gw   = blockIdx.x * 8 + w;
  const int b    = gw >> 8;               // 256 query tiles per batch; 32 blocks/batch
  const int n0   = (gw & 255) * 16;
  const int lm   = lane & 15;
  const int lh   = lane >> 4;
  const float sc = 0.0625f;               // 256^-0.5

  const uint32_t kbase0 = (uint32_t)(uintptr_t)(&kbuf[0][0]);
  const uint32_t kbase1 = (uint32_t)(uintptr_t)(&kbuf[1][0]);
  const uint32_t vbase0 = (uint32_t)(uintptr_t)(&vlds[0][0]);
  const uint32_t vbase1 = (uint32_t)(uintptr_t)(&vlds[1][0]);

  // Issue this thread's share (8 x B128) of one 32-key K+V chunk.
  auto issue_chunk = [&](int m0, uint32_t kdst, uint32_t vdst) {
#pragma unroll
    for (int j = 0; j < 8; ++j) {
      const int u = tid + j * 256;            // unit = 16 bytes, 2048 units
      if (u < 1024) {                         // K: 32 rows x 512B
        const int row = u >> 5;
        const int col = (u & 31) * 16;        // byte offset in row
        const char* g = (const char*)kT +
            (((size_t)b * N_ + m0 + row) * C_) * 2 + col;
        async_b128(kdst + row * 512 + col, g);
      } else {                                // V: 256 rows x 64B
        const int u2  = u - 1024;
        const int c   = u2 >> 2;
        const int seg = (u2 & 3) * 16;
        const char* g = (const char*)vbuf +
            (((size_t)b * C_ + c) * N_ + m0) * 2 + seg;
        async_b128(vdst + c * 64 + seg, g);
      }
    }
  };

  // Q fragments for this wave's 16 queries (held for the whole loop).
  v16h qa[8];
#pragma unroll
  for (int kc = 0; kc < 8; ++kc)
    qa[kc] = load_a16(qT + ((size_t)b * N_ + n0 + lm) * C_ + kc * 32 + lh * 16);

  v8f acc[16];
#pragma unroll
  for (int t = 0; t < 16; ++t) acc[t] = (v8f){};
  float mrun[8], lrun[8];
#pragma unroll
  for (int j = 0; j < 8; ++j) { mrun[j] = -3.0e38f; lrun[j] = 0.f; }

  // Prologue: load chunk 0.
  issue_chunk(0, kbase0, vbase0);
  asm volatile("s_wait_asynccnt 0x0" ::: "memory");
  __syncthreads();

  int cur = 0;
  for (int m0 = 0; m0 < N_; m0 += 32) {
    // Prefetch next chunk into the other buffer.
    if (m0 + 32 < N_)
      issue_chunk(m0 + 32, cur ? kbase0 : kbase1, cur ? vbase0 : vbase1);

    const _Float16* kch = &kbuf[cur][0];
    const _Float16* vch = &vlds[cur][0];

    // ---- scores: two 16x16 tiles over K=256 (16 WMMAs, B from LDS) ----
    v8f s0 = {}, s1 = {};
#pragma unroll
    for (int kc = 0; kc < 8; ++kc) {
      v16h b0 = *(const v16h*)(kch + (size_t)lm * C_ + kc * 32 + lh * 16);
      v16h b1 = *(const v16h*)(kch + (size_t)(16 + lm) * C_ + kc * 32 + lh * 16);
      s0 = WMMA_F16(qa[kc], b0, s0);
      s1 = WMMA_F16(qa[kc], b1, s1);
    }
    // ---- online softmax (D-layout: VGPR j holds rows j / j+8) ----
    float f[8];
#pragma unroll
    for (int j = 0; j < 8; ++j) {
      float t0 = s0[j] * sc, t1 = s1[j] * sc;
      float mx = fmaxf(t0, t1);
#pragma unroll
      for (int d = 8; d >= 1; d >>= 1) mx = fmaxf(mx, __shfl_xor(mx, d, 16));
      const float mnew = fmaxf(mrun[j], mx);
      f[j] = __expf(mrun[j] - mnew);
      t0 = __expf(t0 - mnew);
      t1 = __expf(t1 - mnew);
      float rs = t0 + t1;
#pragma unroll
      for (int d = 8; d >= 1; d >>= 1) rs += __shfl_xor(rs, d, 16);
      lrun[j] = lrun[j] * f[j] + rs;
      mrun[j] = mnew;
      s0[j] = t0; s1[j] = t1;
    }
#pragma unroll
    for (int t = 0; t < 16; ++t)
#pragma unroll
      for (int j = 0; j < 8; ++j) acc[t][j] *= f[j];

    // ---- stage P (D layout) to LDS, reload as A fragment ----
#pragma unroll
    for (int j = 0; j < 8; ++j) {
      const int row = j + lh * 8;
      plds[row * 32 + lm]      = (_Float16)s0[j];
      plds[row * 32 + 16 + lm] = (_Float16)s1[j];
    }
    asm volatile("s_wait_dscnt 0" ::: "memory");   // same-wave DS ordering
    v16h pa = load_a16(plds + lm * 32 + lh * 16);
    asm volatile("" ::: "memory");

    // ---- accumulate P x V^T over 256 channels (16 WMMAs, B from LDS) ----
#pragma unroll
    for (int t = 0; t < 16; ++t) {
      v16h bv_ = *(const v16h*)(vch + (size_t)(t * 16 + lm) * 32 + lh * 16);
      acc[t] = WMMA_F16(pa, bv_, acc[t]);
    }

    // Prefetch must have landed; all waves done reading before buffer swap.
    asm volatile("s_wait_asynccnt 0x0" ::: "memory");
    __syncthreads();
    cur ^= 1;
  }

  float inv[8];
#pragma unroll
  for (int j = 0; j < 8; ++j) inv[j] = 1.f / lrun[j];
#pragma unroll
  for (int t = 0; t < 16; ++t) {
#pragma unroll
    for (int j = 0; j < 8; ++j) {
      const int n = n0 + j + lh * 8;
      const int c = t * 16 + lm;
      oT[((size_t)b * N_ + n) * C_ + c] = (_Float16)(acc[t][j] * inv[j]);
    }
  }
}

// ---------------------------------------------------------------------------
// Projection + residual: out[b,o,n] = x + Wp x o + bp
// ---------------------------------------------------------------------------
__global__ void __launch_bounds__(128) proj_kernel(
    const _Float16* __restrict__ oT, const _Float16* __restrict__ wph,
    const float* __restrict__ bp, const float* __restrict__ x,
    float* __restrict__ out) {
  const int lane = threadIdx.x & 31;
  const int gw   = blockIdx.x * 4 + (threadIdx.x >> 5);
  const int b    = gw >> 12;
  const int r    = gw & 4095;
  const int o0   = (r >> 8) * 16;
  const int n0   = (r & 255) * 16;
  const int lm   = lane & 15;
  const int lh   = lane >> 4;

  v8f acc = {};
#pragma unroll
  for (int kc = 0; kc < 8; ++kc) {
    v16h a  = load_a16(wph + (size_t)(o0 + lm) * C_ + kc * 32 + lh * 16);
    v16h bb = *(const v16h*)(oT + ((size_t)b * N_ + n0 + lm) * C_ + kc * 32 + lh * 16);
    acc = WMMA_F16(a, bb, acc);
  }
#pragma unroll
  for (int j = 0; j < 8; ++j) {
    const int o = o0 + j + lh * 8;
    const size_t idx = ((size_t)b * C_ + o) * N_ + n0 + lm;
    out[idx] = x[idx] + acc[j] + bp[o];
  }
}

// ---------------------------------------------------------------------------
extern "C" void kernel_launch(void* const* d_in, const int* in_sizes, int n_in,
                              void* d_out, int out_size, void* d_ws, size_t ws_size,
                              hipStream_t stream) {
  (void)in_sizes; (void)n_in; (void)out_size; (void)ws_size;
  const float* x   = (const float*)d_in[0];
  const float* gns = (const float*)d_in[1];
  const float* gnb = (const float*)d_in[2];
  const float* wq  = (const float*)d_in[3];
  const float* bq  = (const float*)d_in[4];
  const float* wk  = (const float*)d_in[5];
  const float* bk  = (const float*)d_in[6];
  const float* wv  = (const float*)d_in[7];
  const float* bv  = (const float*)d_in[8];
  const float* wp  = (const float*)d_in[9];
  const float* bp  = (const float*)d_in[10];
  float* out = (float*)d_out;

  char* ws = (char*)d_ws;
  const size_t TS = (size_t)B_ * N_ * C_ * sizeof(_Float16);   // 16 MB per tensor
  _Float16* hnT  = (_Float16*)(ws);
  _Float16* qT   = (_Float16*)(ws + 1 * TS);
  _Float16* kT   = (_Float16*)(ws + 2 * TS);
  _Float16* vbuf = (_Float16*)(ws + 3 * TS);
  _Float16* oT   = (_Float16*)(ws + 4 * TS);
  _Float16* wh   = (_Float16*)(ws + 5 * TS);                   // 4 x 128 KB
  const size_t WM = (size_t)C_ * C_;

  wcvt_kernel<<<(C_ * C_ + 255) / 256, 256, 0, stream>>>(wq, wk, wv, wp, wh);
  gn_kernel<<<B_ * GROUPS_, 256, 0, stream>>>(x, gns, gnb, hnT);
  qkv_kernel<<<8192, 128, 0, stream>>>(hnT, wh, wh + WM, wh + 2 * WM,
                                       bq, bk, bv, qT, kT, vbuf);
  attn_kernel<<<256, 256, 0, stream>>>(qT, kT, vbuf, oT);
  proj_kernel<<<8192, 128, 0, stream>>>(oT, wh + 3 * WM, bp, x, out);
}
